// Attention_12704513261700
// MI455X (gfx1250) — matrix operand order repaired and checked by reading, not compiled, not run on hardware
//
#include <hip/hip_runtime.h>

// ---------------------------------------------------------------------------
// CDNA5 (gfx1250) fused windowed-attention.  wave32, WMMA f16 16x16x32.
// ---------------------------------------------------------------------------

typedef _Float16 half16 __attribute__((ext_vector_type(16)));
typedef float    float8 __attribute__((ext_vector_type(8)));

union V16H { half16 v; uint4 q[2]; };

// A-operand: lane holds two separated 8-half (16B) runs: [0..7] and [+16..+23]
__device__ __forceinline__ half16 ld16_sep(const _Float16* p) {
    V16H u;
    u.q[0] = *(const uint4*)(p);
    u.q[1] = *(const uint4*)(p + 16);
    return u.v;
}
// B-operand: lane holds two separated 8-half (16B) runs: [0..7] and [+16..+23]
__device__ __forceinline__ half16 ld16_cont(const _Float16* p) {
    V16H u;
    u.q[0] = *(const uint4*)(p);
    u.q[1] = *(const uint4*)(p + 16);
    return u.v;
}

#define WMMA(a, b, c) \
    __builtin_amdgcn_wmma_f32_16x16x32_f16(false, (a), false, (b), (short)0, (c), false, false)

// ---------------- LDS layout (bytes) ----------------
// live windows chosen so overlays are safe across __syncthreads() phases
#define LDS_XLN   0         // f16 [64 pix][136]          17408
#define LDS_QS    17408     // f16 [4][64 i][40]          20480
#define LDS_KS    37888     // f16 [4][64 j][40]          20480
#define LDS_VT    58368     // f16 [4][32 dh][72]         18432
#define LDS_SIM   76800     // f32 [4][64 i][65]          66560
#define LDS_BIAS  143360    // f32 [289]                   1168
#define LDS_PSUM  144528    // f32 [4][64]                 1024
#define LDS_PSQ   145552    // f32 [4][64]                 1024
#define LDS_PMEAN 146576    // f32 [64]                     256
#define LDS_PINV  146832    // f32 [64]                     256
#define LDS_TOTAL 147088
#define LDS_ATTN  0         // f16 [4][64 i][72]  (overlay xln+Qs, dead then)
#define LDS_OUTB  37888     // f16 [64 pix][136]  (overlay Ks, dead then)

#define SCALE 0.17677669529663689f  // 32^-0.5

// ---------------------------------------------------------------------------
// prep 1: fp32 -> f16 weight conversion into workspace
// ---------------------------------------------------------------------------
__global__ void prep_weights(const float* __restrict__ wqkv, const float* __restrict__ wout,
                             _Float16* __restrict__ wqkv_h, _Float16* __restrict__ wout_h) {
    int i = blockIdx.x * 256 + threadIdx.x;
    if (i < 49152) wqkv_h[i] = (_Float16)wqkv[i];
    if (i < 16384) wout_h[i] = (_Float16)wout[i];
}

// ---------------------------------------------------------------------------
// prep 2: dynamic-position-bias MLP over the 289 (dy,dx) coords
// ---------------------------------------------------------------------------
__global__ void prep_dpb(const float* __restrict__ w1, const float* __restrict__ b1,
                         const float* __restrict__ g1, const float* __restrict__ bb1,
                         const float* __restrict__ w2, const float* __restrict__ b2,
                         const float* __restrict__ g2, const float* __restrict__ bb2,
                         const float* __restrict__ w3, const float* __restrict__ b3,
                         float* __restrict__ biases) {
    int t = threadIdx.x;
    if (t >= 289) return;
    float dy = (float)(t / 17 - 8);
    float dx = (float)(t % 17 - 8);

    float h1[32], h2[32];
    float s = 0.f;
    for (int k = 0; k < 32; ++k) { h1[k] = w1[2*k]*dy + w1[2*k+1]*dx + b1[k]; s += h1[k]; }
    float m = s * (1.f/32.f), v = 0.f;
    for (int k = 0; k < 32; ++k) { float d = h1[k]-m; v += d*d; }
    float r = rsqrtf(v*(1.f/32.f) + 1e-5f);
    for (int k = 0; k < 32; ++k) h1[k] = fmaxf((h1[k]-m)*r*g1[k] + bb1[k], 0.f);

    s = 0.f;
    for (int k = 0; k < 32; ++k) {
        float acc = b2[k];
        for (int j = 0; j < 32; ++j) acc += w2[k*32+j]*h1[j];
        h2[k] = acc; s += acc;
    }
    m = s * (1.f/32.f); v = 0.f;
    for (int k = 0; k < 32; ++k) { float d = h2[k]-m; v += d*d; }
    r = rsqrtf(v*(1.f/32.f) + 1e-5f);
    float outv = b3[0];
    for (int k = 0; k < 32; ++k) {
        float h = fmaxf((h2[k]-m)*r*g2[k] + bb2[k], 0.f);
        outv += w3[k]*h;
    }
    biases[t] = outv;
}

// ---------------------------------------------------------------------------
// main fused kernel: one workgroup (256 thr = 8 waves) per 8x8 window
// ---------------------------------------------------------------------------
__global__ __launch_bounds__(256)
void attn_fused(const float* __restrict__ x,
                const float* __restrict__ ln_g, const float* __restrict__ ln_b,
                const _Float16* __restrict__ wqkv_h, const _Float16* __restrict__ wout_h,
                const float* __restrict__ b_out, const float* __restrict__ bias_ws,
                float* __restrict__ out) {
    __shared__ __attribute__((aligned(16))) char smem[LDS_TOTAL];

    _Float16* xln  = (_Float16*)(smem + LDS_XLN);
    _Float16* Qs   = (_Float16*)(smem + LDS_QS);
    _Float16* Ks   = (_Float16*)(smem + LDS_KS);
    _Float16* Vt   = (_Float16*)(smem + LDS_VT);
    _Float16* attnL= (_Float16*)(smem + LDS_ATTN);
    _Float16* OutB = (_Float16*)(smem + LDS_OUTB);
    float* simS  = (float*)(smem + LDS_SIM);
    float* biasL = (float*)(smem + LDS_BIAS);
    float* psum  = (float*)(smem + LDS_PSUM);
    float* psq   = (float*)(smem + LDS_PSQ);
    float* pmean = (float*)(smem + LDS_PMEAN);
    float* pinv  = (float*)(smem + LDS_PINV);

    const int tid  = threadIdx.x;
    const int lane = tid & 31;
    const int wave = tid >> 5;
    const int l15  = lane & 15;
    const int hsel = lane >> 4;     // 0: lanes 0-15, 1: lanes 16-31

    const int blk = blockIdx.x;
    const int b  = blk >> 10;
    const int wh = (blk >> 5) & 31;
    const int ww = blk & 31;
    const int winBase = b*128*65536 + wh*8*256 + ww*8;

    // bias table -> LDS
    for (int i = tid; i < 289; i += 256) biasL[i] = bias_ws[i];

    // ---- Phase A: channel-LN statistics (pixel p, channel group g) --------
    const int p = tid & 63;
    const int g = tid >> 6;
    const float* xp = x + winBase + (p >> 3)*256 + (p & 7);
    {
        float s = 0.f, sq = 0.f;
        for (int d = g*32; d < g*32 + 32; ++d) {
            float v = xp[d << 16];
            s += v; sq += v*v;
        }
        psum[g*64 + p] = s;
        psq [g*64 + p] = sq;
    }
    __syncthreads();
    if (tid < 64) {
        float s  = psum[tid] + psum[64+tid] + psum[128+tid] + psum[192+tid];
        float sq = psq [tid] + psq [64+tid] + psq [128+tid] + psq [192+tid];
        float m  = s * (1.f/128.f);
        float v  = sq * (1.f/128.f) - m*m;
        pmean[tid] = m;
        pinv [tid] = 1.f / (v + 1e-5f);   // reference divides by (v+eps), no sqrt
    }
    __syncthreads();

    // ---- Phase B: x_ln -> f16 LDS [pixel][chan] ---------------------------
    {
        float m = pmean[p], inv = pinv[p];
        for (int d = g*32; d < g*32 + 32; ++d) {
            float v = xp[d << 16];
            xln[p*136 + d] = (_Float16)((v - m) * inv * ln_g[d] + ln_b[d]);
        }
    }
    __syncthreads();

    // ---- Phase C: QKV GEMM (384x128 @ 128x64), scatter into Q/K/Vt -------
    // mm = 0 -> Q rows (e in [0,128)), mm = 1 -> K, mm = 2 -> V: segment is a
    // compile-time constant after unrolling, so the scatter is branch-free.
    #pragma unroll
    for (int mm = 0; mm < 3; ++mm) {
        const int mtile = wave + 8*mm;
        const int erow  = mtile*16 + l15;          // global weight row
        const int ebase = wave*16 + hsel*8;        // e within segment, r=0
        #pragma unroll
        for (int ntile = 0; ntile < 4; ++ntile) {
            float8 acc = {0.f,0.f,0.f,0.f,0.f,0.f,0.f,0.f};
            #pragma unroll
            for (int k0 = 0; k0 < 4; ++k0) {
                half16 a  = ld16_sep (wqkv_h + erow*128 + k0*32 + hsel*8);
                half16 bm = ld16_cont(xln + (ntile*16 + l15)*136 + k0*32 + hsel*8);
                acc = WMMA(a, bm, acc);
            }
            const int pcol = ntile*16 + l15;
            const int phi  = pcol >> 5;            // 0/1: which 32-pixel half
            const int plo  = pcol & 31;            // dh
            if (mm == 0) {                         // Q (scaled)
                #pragma unroll
                for (int r = 0; r < 8; ++r) {
                    int e2 = ebase + r;
                    int head = e2 >> 5, i = 2*(e2 & 31) + phi;
                    Qs[(head*64 + i)*40 + plo] = (_Float16)(acc[r] * SCALE);
                }
            } else if (mm == 1) {                  // K
                #pragma unroll
                for (int r = 0; r < 8; ++r) {
                    int e2 = ebase + r;
                    int head = e2 >> 5, i = 2*(e2 & 31) + phi;
                    Ks[(head*64 + i)*40 + plo] = (_Float16)acc[r];
                }
            } else {                               // V transposed: Vt[head][dh][j]
                #pragma unroll
                for (int r = 0; r < 8; ++r) {
                    int e2 = ebase + r;
                    int head = e2 >> 5, j = 2*(e2 & 31) + phi;
                    Vt[(head*32 + plo)*72 + j] = (_Float16)acc[r];
                }
            }
        }
    }
    __syncthreads();

    // ---- Phase D: sim = Q' K'^T  (4 heads x 4x4 tiles, K=32) -------------
    #pragma unroll
    for (int t = 0; t < 8; ++t) {
        int gid = wave*8 + t;
        int head = gid >> 4, rem = gid & 15;
        int mi = rem >> 2, nj = rem & 3;
        half16 a  = ld16_sep (Qs + (head*64 + mi*16 + l15)*40 + hsel*8);
        half16 bm = ld16_cont(Ks + (head*64 + nj*16 + l15)*40 + hsel*8);
        float8 acc = {0.f,0.f,0.f,0.f,0.f,0.f,0.f,0.f};
        acc = WMMA(a, bm, acc);
        float* dst = simS + (head*64 + mi*16 + hsel*8)*65 + nj*16 + l15;
        #pragma unroll
        for (int r = 0; r < 8; ++r) dst[r*65] = acc[r];
    }
    __syncthreads();

    // ---- Phase E: + rel_bias, softmax, -> f16 attn ------------------------
    {
        int head = tid >> 6, i = tid & 63;
        int yi = i >> 3, xi = i & 7;
        float* srow = simS + (head*64 + i)*65;
        float mx = -1e30f;
        for (int j = 0; j < 64; ++j) {
            int idx = (yi - (j >> 3) + 7)*15 + (xi - (j & 7) + 7);  // faithful 15-stride
            float v = srow[j] + biasL[idx];
            srow[j] = v;
            mx = fmaxf(mx, v);
        }
        float sum = 0.f;
        for (int j = 0; j < 64; ++j) {
            float e = __expf(srow[j] - mx);
            srow[j] = e; sum += e;
        }
        float rs = 1.f / sum;
        _Float16* arow = attnL + (head*64 + i)*72;
        for (int j = 0; j < 64; ++j) arow[j] = (_Float16)(srow[j] * rs);
    }
    __syncthreads();

    // ---- Phase F: out = attn @ V'  -> OutB[pixel][head*32+dh] ------------
    #pragma unroll
    for (int t = 0; t < 4; ++t) {
        int gid = wave*4 + t;
        int head = gid >> 3, rem = gid & 7;
        int mi = rem >> 1, nd = rem & 1;
        float8 acc = {0.f,0.f,0.f,0.f,0.f,0.f,0.f,0.f};
        #pragma unroll
        for (int k0 = 0; k0 < 2; ++k0) {
            half16 a  = ld16_sep (attnL + (head*64 + mi*16 + l15)*72 + k0*32 + hsel*8);
            half16 bm = ld16_cont(Vt    + (head*32 + nd*16 + l15)*72 + k0*32 + hsel*8);
            acc = WMMA(a, bm, acc);
        }
        int dh = nd*16 + l15;
        _Float16* dst = OutB + (mi*16 + hsel*8)*136 + head*32 + dh;
        #pragma unroll
        for (int r = 0; r < 8; ++r) dst[r*136] = (_Float16)acc[r];
    }
    __syncthreads();

    // ---- Phase G: final projection 128x128 @ 128x64, + b_out, store ------
    {
        const int mtile = wave;
        #pragma unroll
        for (int ntile = 0; ntile < 4; ++ntile) {
            float8 acc = {0.f,0.f,0.f,0.f,0.f,0.f,0.f,0.f};
            #pragma unroll
            for (int k0 = 0; k0 < 4; ++k0) {
                half16 a  = ld16_sep (wout_h + (mtile*16 + l15)*128 + k0*32 + hsel*8);
                half16 bm = ld16_cont(OutB + (ntile*16 + l15)*136 + k0*32 + hsel*8);
                acc = WMMA(a, bm, acc);
            }
            int pcol = ntile*16 + l15;
            int y = pcol >> 3, xc = pcol & 7;
            float* dst = out + winBase + (mtile*16 + hsel*8)*65536 + y*256 + xc;
            const float* bo = b_out + mtile*16 + hsel*8;
            #pragma unroll
            for (int r = 0; r < 8; ++r) dst[r*65536] = acc[r] + bo[r];
        }
    }
}

// ---------------------------------------------------------------------------
extern "C" void kernel_launch(void* const* d_in, const int* in_sizes, int n_in,
                              void* d_out, int out_size, void* d_ws, size_t ws_size,
                              hipStream_t stream) {
    (void)in_sizes; (void)n_in; (void)out_size; (void)ws_size;
    const float* x     = (const float*)d_in[0];
    const float* ln_g  = (const float*)d_in[1];
    const float* ln_b  = (const float*)d_in[2];
    const float* w_qkv = (const float*)d_in[3];
    const float* w_out = (const float*)d_in[4];
    const float* b_out = (const float*)d_in[5];
    const float* dw1 = (const float*)d_in[6];
    const float* db1 = (const float*)d_in[7];
    const float* dg1 = (const float*)d_in[8];
    const float* dbb1= (const float*)d_in[9];
    const float* dw2 = (const float*)d_in[10];
    const float* db2 = (const float*)d_in[11];
    const float* dg2 = (const float*)d_in[12];
    const float* dbb2= (const float*)d_in[13];
    const float* dw3 = (const float*)d_in[14];
    const float* db3 = (const float*)d_in[15];
    float* out = (float*)d_out;

    _Float16* wqkv_h = (_Float16*)d_ws;                       // 98304 B
    _Float16* wout_h = (_Float16*)((char*)d_ws + 98304);      // 32768 B
    float*    bias_ws= (float*)   ((char*)d_ws + 131072);     //  1156 B

    prep_weights<<<192, 256, 0, stream>>>(w_qkv, w_out, wqkv_h, wout_h);
    prep_dpb<<<1, 320, 0, stream>>>(dw1, db1, dg1, dbb1, dw2, db2, dg2, dbb2, dw3, db3, bias_ws);
    attn_fused<<<4096, 256, 0, stream>>>(x, ln_g, ln_b, wqkv_h, wout_h, b_out, bias_ws, out);
}
